// LightningGQABlock_29841432772786
// MI455X (gfx1250) — compile-verified, matrix operand order checked by reading, not run on hardware
//
#include <hip/hip_runtime.h>
#include <math.h>

// ---------------- problem constants ----------------
#define S_LEN   2048
#define B_SZ    2
#define D_MODEL 1024
#define HKV     4
#define G_GRP   4
#define DH      64
#define WIN     256
#define QKVN    1536     // D + 2*HKV*DH
#define KOFF    1024
#define VOFF    1280
#define DFF     4096
#define MROWS   (B_SZ * S_LEN)   // 4096

typedef unsigned short u16;
typedef unsigned int   u32;
typedef unsigned long long u64;
typedef __attribute__((ext_vector_type(16))) __bf16 v16bf;
typedef __attribute__((ext_vector_type(8)))  float  v8f;
typedef __attribute__((ext_vector_type(4)))  unsigned int u32x4;
typedef __attribute__((ext_vector_type(8)))  int i32x8;
typedef __attribute__((ext_vector_type(4)))  int i32x4;

// ---------------- small helpers ----------------
__device__ __forceinline__ u16 f2bf(float f) {
  u32 u = __builtin_bit_cast(u32, f);
  u += 0x7fffu + ((u >> 16) & 1u);         // round to nearest even
  return (u16)(u >> 16);
}

__device__ __forceinline__ v8f v8f_zero() {
  v8f z;
#pragma unroll
  for (int i = 0; i < 8; ++i) z[i] = 0.0f;
  return z;
}

union FragU { v16bf v; uint4 q[2]; u16 u[16]; };

// CDNA5 16-bit A/B fragment layout (wave32): lane = row(A)/col(B) (lane&15),
// K chunks split by lane half: elems[0..7]=K(half*8+i), elems[8..15]=K(16+half*8+i).
// base32 points at this lane's 32-element (64B, 16B-aligned) K-row.
__device__ __forceinline__ v16bf load_frag(const u16* base32, int hf) {
  FragU f;
  const uint4* p = (const uint4*)base32;
  f.q[0] = p[hf];
  f.q[1] = p[2 + hf];
  return f.v;
}

#define WMMA_BF16(A, B, C) \
  __builtin_amdgcn_wmma_f32_16x16x32_bf16(false, (A), false, (B), (short)0, (C), false, false)

// ---------------- Tensor Data Mover: 2-D bf16 tile -> LDS ----------------
// Builds a D# per CDNA5 ISA and issues TENSOR_LOAD_TO_LDS.
// tile_w elems per row (row-major, contiguous), tile_h rows, row stride in elems.
__device__ __forceinline__ void tdm_load_2d_bf16(const u16* gptr, u32 lds_byte_off,
                                                 int tile_w, int tile_h,
                                                 int tensor_w, int tensor_h,
                                                 int row_stride_elems) {
  u64 ga = (u64)(uintptr_t)gptr;
  u32x4 g0;
  g0[0] = 1u;                                         // count=1, user descriptor
  g0[1] = lds_byte_off;                               // lds_addr (bytes)
  g0[2] = (u32)(ga & 0xffffffffu);                    // global_addr[31:0]
  g0[3] = (u32)((ga >> 32) & 0x1ffffffu) | (2u << 30);// global_addr[56:32] | type=2
  i32x8 g1;
  g1[0] = (1 << 16);                                  // data_size=1 -> 2 bytes/elem
  g1[1] = (int)((u32)(tensor_w & 0xffff) << 16);      // [63:48] tensor_dim0 lo
  g1[2] = (int)(((u32)tensor_w >> 16) & 0xffffu) |    // [79:64] tensor_dim0 hi
          (int)((u32)(tensor_h & 0xffff) << 16);      // [95:80] tensor_dim1 lo
  g1[3] = (int)(((u32)tensor_h >> 16) & 0xffffu) |    // [111:96] tensor_dim1 hi
          (int)((u32)(tile_w & 0xffff) << 16);        // [127:112] tile_dim0
  g1[4] = (int)(tile_h & 0xffff);                     // [143:128] tile_dim1 (tile_dim2=0)
  g1[5] = row_stride_elems;                           // [191:160] tensor_dim0_stride lo
  g1[6] = 0;                                          // stride hi / dim1_stride lo
  g1[7] = 0;
  i32x4 z4 = {0, 0, 0, 0};                            // groups 2/3 unused (2-D tensor)
#if __has_include(<hip/amd_detail/amd_gfx1250_TDM.h>)
  i32x8 z8 = {0, 0, 0, 0, 0, 0, 0, 0};
  __builtin_amdgcn_tensor_load_to_lds(g0, g1, z4, z4, z8, 0);   // therock 6-arg form
#else
  __builtin_amdgcn_tensor_load_to_lds(g0, g1, z4, z4, 0);       // ROCm 7.2 5-arg form
#endif
}

// ---------------- fp32 -> bf16 cast ----------------
__global__ __launch_bounds__(256) void cast_f32_bf16(const float* __restrict__ in,
                                                     u16* __restrict__ out, int n) {
  int i = blockIdx.x * 256 + threadIdx.x;
  if (i < n) out[i] = f2bf(in[i]);
}

// ---------------- RMSNorm (fp32 in, bf16 out) ----------------
__global__ __launch_bounds__(256) void rmsnorm_kernel(const float* __restrict__ x,
                                                      const float* __restrict__ w,
                                                      u16* __restrict__ out) {
  const int row = blockIdx.x;
  const float* xr = x + (size_t)row * D_MODEL;
  const int j = threadIdx.x * 4;
  float4 xv = *(const float4*)(xr + j);
  float ss = xv.x * xv.x + xv.y * xv.y + xv.z * xv.z + xv.w * xv.w;
#pragma unroll
  for (int off = 1; off < 32; off <<= 1) ss += __shfl_xor(ss, off, 32);
  __shared__ float red[8];
  if ((threadIdx.x & 31) == 0) red[threadIdx.x >> 5] = ss;
  __syncthreads();
  ss = 0.0f;
#pragma unroll
  for (int i = 0; i < 8; ++i) ss += red[i];
  const float inv = rsqrtf(ss * (1.0f / D_MODEL) + 1e-6f);
  const float4 wv = *(const float4*)(w + j);
  u16* orow = out + (size_t)row * D_MODEL + j;
  orow[0] = f2bf(xv.x * inv * wv.x);
  orow[1] = f2bf(xv.y * inv * wv.y);
  orow[2] = f2bf(xv.z * inv * wv.z);
  orow[3] = f2bf(xv.w * inv * wv.w);
}

// ---------------- generic bf16 WMMA GEMM ----------------
// C[MxN] = A[MxK] * B[KxN], A/B bf16 row-major.  Block tile 128x128, 8 waves,
// wave tile 32x64 (2 A-frags x 4 B-frags = 8 WMMAs per 32-K step).
// A tile is staged by the Tensor Data Mover (double buffered, TENSORcnt),
// B tile is staged transposed through VGPRs (TDM cannot transpose).
// EPI: 0 = f32 out (+optional residual)       (out-proj -> x1)
//      1 = bf16 out, bias + exact GELU        (FFN1)
//      2 = f32 out, bias + residual           (FFN2 -> d_out)
//      3 = bf16 out, plain                    (QKV)
template <int EPI>
__global__ __launch_bounds__(256) void gemm_bf16_kernel(
    const u16* __restrict__ A, const u16* __restrict__ Bm,
    float* __restrict__ Cf, u16* __restrict__ Cb,
    const float* __restrict__ bias, const float* __restrict__ resid,
    int M, int N, int K) {
  __shared__ u16 Alds[2][128 * 32];   // 8KB x2 : [row][k] row-major (TDM target)
  __shared__ u16 BTl[2][128 * 32];    // 8KB x2 : [col][k] transposed

  const int tid  = threadIdx.x;
  const int lane = tid & 31, wave = tid >> 5;
  const int wm = wave >> 1, wn = wave & 1;      // 4x2 wave grid -> 32x64 per wave
  const int m0 = blockIdx.y * 128, n0 = blockIdx.x * 128;
  const int hf = lane >> 4, ln = lane & 15;
  const int ksteps = K >> 5;

  v8f acc0[4], acc1[4];
#pragma unroll
  for (int j = 0; j < 4; ++j) { acc0[j] = v8f_zero(); acc1[j] = v8f_zero(); }

  auto stage = [&](int k0, int buf) {
    if (wave == 0) {
      // A tile: 128 rows x 32 K elems, row stride K -> LDS row-major, via TDM
      tdm_load_2d_bf16(A + (size_t)m0 * K + k0,
                       (u32)(uintptr_t)&Alds[buf][0],
                       /*tile_w=*/32, /*tile_h=*/128,
                       /*tensor_w=*/K, /*tensor_h=*/M, /*row_stride=*/K);
    }
    // B tile transposed: 32 K rows x 128 cols -> BT[col][k]
    for (int idx = tid; idx < 4096; idx += 256) {
      int kr = idx >> 7, n = idx & 127;
      BTl[buf][n * 32 + kr] = Bm[(size_t)(k0 + kr) * N + n0 + n];
    }
  };

  stage(0, 0);
  for (int it = 0; it < ksteps; ++it) {
    const int buf = it & 1;
    __builtin_amdgcn_s_wait_tensorcnt(0);   // TDM for `buf` complete (no-op on waves 1..7)
    __syncthreads();                        // ..and B-tile stores visible to all waves
    if (it + 1 < ksteps) stage((it + 1) << 5, buf ^ 1);  // overlap next stage with compute

    v16bf a0 = load_frag(&Alds[buf][(wm * 32 + ln) * 32], hf);
    v16bf a1 = load_frag(&Alds[buf][(wm * 32 + 16 + ln) * 32], hf);
#pragma unroll
    for (int j = 0; j < 4; ++j) {
      v16bf b = load_frag(&BTl[buf][(wn * 64 + j * 16 + ln) * 32], hf);
      acc0[j] = WMMA_BF16(a0, b, acc0[j]);
      acc1[j] = WMMA_BF16(a1, b, acc1[j]);
    }
  }

  // epilogue: C layout -> row m = hf*8+r, col n = ln (per 16x16 tile)
#pragma unroll
  for (int ti = 0; ti < 2; ++ti)
#pragma unroll
    for (int j = 0; j < 4; ++j) {
      v8f accv = (ti == 0) ? acc0[j] : acc1[j];
#pragma unroll
      for (int r = 0; r < 8; ++r) {
        int m = m0 + wm * 32 + ti * 16 + hf * 8 + r;
        int n = n0 + wn * 64 + j * 16 + ln;
        size_t o = (size_t)m * N + n;
        float v = accv[r];
        if (EPI == 0) {
          if (resid) v += resid[o];
          Cf[o] = v;
        } else if (EPI == 1) {
          v += bias[n];
          v = 0.5f * v * (1.0f + erff(v * 0.70710678118f));  // exact GELU
          Cb[o] = f2bf(v);
        } else if (EPI == 2) {
          v += bias[n] + resid[o];
          Cf[o] = v;
        } else {
          Cb[o] = f2bf(v);
        }
      }
    }
}

// ---------------- sliding-window GQA flash attention ----------------
// grid: (S/16 query tiles, HKV, B); block: 128 threads = 4 waves = the 4 query
// groups of one KV head, sharing K/V tiles in LDS.
__global__ __launch_bounds__(128) void attn_kernel(const u16* __restrict__ qkv,
                                                   u16* __restrict__ out) {
  const int qt = blockIdx.x, h = blockIdx.y, b = blockIdx.z;
  const int wave = threadIdx.x >> 5;  // query group g
  const int lane = threadIdx.x & 31;
  const int hf = lane >> 4, ln = lane & 15;
  const int q0 = qt * 16;

  __shared__ u16 Kt[32 * 64];        // [key][dh]        (QK^T B-frags)
  __shared__ u16 Vt[64 * 32];        // [dh][key] transp (PV   B-frags)
  __shared__ u16 Pl[4][16 * 32];     // per-wave P tile re-layout

  // Q fragments: 16x64 tile -> two 16x32 A-frags (pre-converted bf16 in qkv)
  const u16* qrow = qkv + ((size_t)(b * S_LEN) + q0 + ln) * QKVN + (h * G_GRP + wave) * DH;
  v16bf qf0, qf1;
  {
    FragU f;
    const uint4* p = (const uint4*)qrow;
    f.q[0] = p[hf];     f.q[1] = p[2 + hf]; qf0 = f.v;   // dh 0..31
    f.q[0] = p[4 + hf]; f.q[1] = p[6 + hf]; qf1 = f.v;   // dh 32..63
  }

  float Mrow[8], Lrow[8];
  v8f O0 = v8f_zero(), O1 = v8f_zero(), O2 = v8f_zero(), O3 = v8f_zero();
#pragma unroll
  for (int r = 0; r < 8; ++r) { Mrow[r] = -1e28f; Lrow[r] = 0.0f; }

  const float scale = 0.125f;  // 1/sqrt(DH)
  int kb0 = q0 - (WIN - 1);
  if (kb0 < 0) kb0 = 0;
  kb0 &= ~31;

  for (int kb = kb0; kb <= q0 + 15; kb += 32) {
    __syncthreads();
    // stage K tile [32][64]
    for (int c = threadIdx.x; c < 256; c += 128) {
      int kr = c >> 3, seg = c & 7;
      ((uint4*)Kt)[c] =
          *(const uint4*)(qkv + ((size_t)(b * S_LEN) + kb + kr) * QKVN + KOFF + h * DH + seg * 8);
    }
    // stage V tile transposed [64][32]
    for (int idx = threadIdx.x; idx < 2048; idx += 128) {
      int kr = idx >> 6, d = idx & 63;
      Vt[d * 32 + kr] = qkv[((size_t)(b * S_LEN) + kb + kr) * QKVN + VOFF + h * DH + d];
    }
    __syncthreads();

    // scores: S[16 x 32] over the 32-key block (two 16x16 WMMA tiles, K=64 contraction)
    v8f s0 = v8f_zero(), s1 = v8f_zero();
    {
      v16bf k00 = load_frag(&Kt[ln * 64 + 0], hf);
      v16bf k01 = load_frag(&Kt[ln * 64 + 32], hf);
      v16bf k10 = load_frag(&Kt[(16 + ln) * 64 + 0], hf);
      v16bf k11 = load_frag(&Kt[(16 + ln) * 64 + 32], hf);
      s0 = WMMA_BF16(qf0, k00, s0);
      s0 = WMMA_BF16(qf1, k01, s0);
      s1 = WMMA_BF16(qf0, k10, s1);
      s1 = WMMA_BF16(qf1, k11, s1);
    }

    // online softmax; row m = hf*8+r lives across this half's 16 lanes (col = ln)
#pragma unroll
    for (int r = 0; r < 8; ++r) {
      int q = q0 + hf * 8 + r;
      int key0 = kb + ln, key1 = kb + 16 + ln;
      float v0 = s0[r] * scale, v1 = s1[r] * scale;
      s0[r] = (q >= key0 && (q - key0) < WIN) ? v0 : -1e30f;
      s1[r] = (q >= key1 && (q - key1) < WIN) ? v1 : -1e30f;
      float bm = fmaxf(s0[r], s1[r]);
#pragma unroll
      for (int off = 1; off < 16; off <<= 1) bm = fmaxf(bm, __shfl_xor(bm, off, 16));
      float mn = fmaxf(Mrow[r], bm);
      float alpha = __expf(Mrow[r] - mn);
      Mrow[r] = mn;
      float p0 = __expf(s0[r] - mn), p1 = __expf(s1[r] - mn);
      s0[r] = p0; s1[r] = p1;
      float rs = p0 + p1;
#pragma unroll
      for (int off = 1; off < 16; off <<= 1) rs += __shfl_xor(rs, off, 16);
      Lrow[r] = Lrow[r] * alpha + rs;
      O0[r] *= alpha; O1[r] *= alpha; O2[r] *= alpha; O3[r] *= alpha;
    }

    // P (C-layout) -> LDS -> A-frag layout
    u16* P = Pl[wave];
#pragma unroll
    for (int r = 0; r < 8; ++r) {
      int mrow = hf * 8 + r;
      P[mrow * 32 + ln] = f2bf(s0[r]);
      P[mrow * 32 + 16 + ln] = f2bf(s1[r]);
    }
    __syncthreads();

    // O += P @ V  (contraction over 32 keys; 4 dh chunks of 16)
    v16bf pf = load_frag(&P[ln * 32], hf);
    v16bf v0 = load_frag(&Vt[(0 + ln) * 32], hf);
    v16bf v1 = load_frag(&Vt[(16 + ln) * 32], hf);
    v16bf v2 = load_frag(&Vt[(32 + ln) * 32], hf);
    v16bf v3 = load_frag(&Vt[(48 + ln) * 32], hf);
    O0 = WMMA_BF16(pf, v0, O0);
    O1 = WMMA_BF16(pf, v1, O1);
    O2 = WMMA_BF16(pf, v2, O2);
    O3 = WMMA_BF16(pf, v3, O3);
  }

  // normalize and store bf16 into attn buffer [B,S,D]
#pragma unroll
  for (int r = 0; r < 8; ++r) {
    float inv = 1.0f / Lrow[r];
    int q = q0 + hf * 8 + r;
    u16* orow = out + ((size_t)(b * S_LEN) + q) * D_MODEL + (h * G_GRP + wave) * DH;
    orow[0 + ln]  = f2bf(O0[r] * inv);
    orow[16 + ln] = f2bf(O1[r] * inv);
    orow[32 + ln] = f2bf(O2[r] * inv);
    orow[48 + ln] = f2bf(O3[r] * inv);
  }
}

// ---------------- launch ----------------
extern "C" void kernel_launch(void* const* d_in, const int* in_sizes, int n_in,
                              void* d_out, int out_size, void* d_ws, size_t ws_size,
                              hipStream_t stream) {
  const float* x       = (const float*)d_in[0];
  const float* w_qkv   = (const float*)d_in[1];
  const float* w_out   = (const float*)d_in[2];
  const float* norm1_w = (const float*)d_in[3];
  const float* norm2_w = (const float*)d_in[4];
  const float* w_ffn1  = (const float*)d_in[5];
  const float* b_ffn1  = (const float*)d_in[6];
  const float* w_ffn2  = (const float*)d_in[7];
  const float* b_ffn2  = (const float*)d_in[8];
  float* outp = (float*)d_out;

  // workspace carve-up (~110 MB total)
  char* ws = (char*)d_ws;
  size_t off = 0;
  auto carve = [&](size_t bytes) {
    void* p = ws + off;
    off += (bytes + 255) & ~(size_t)255;
    return p;
  };
  u16* wqkv_bf  = (u16*)carve((size_t)D_MODEL * QKVN * 2);
  u16* wout_bf  = (u16*)carve((size_t)D_MODEL * D_MODEL * 2);
  u16* wffn1_bf = (u16*)carve((size_t)D_MODEL * DFF * 2);
  u16* wffn2_bf = (u16*)carve((size_t)DFF * D_MODEL * 2);
  u16* hA       = (u16*)carve((size_t)MROWS * D_MODEL * 2);
  u16* qkv_bf   = (u16*)carve((size_t)MROWS * QKVN * 2);
  u16* attn_bf  = (u16*)carve((size_t)MROWS * D_MODEL * 2);
  float* x1     = (float*)carve((size_t)MROWS * D_MODEL * 4);
  u16* h2       = (u16*)carve((size_t)MROWS * D_MODEL * 2);
  u16* ffn1_bf  = (u16*)carve((size_t)MROWS * DFF * 2);
  (void)ws_size; (void)in_sizes; (void)n_in; (void)out_size;

  // weight casts (L2-resident afterwards)
  auto cast = [&](const float* src, u16* dst, int n) {
    cast_f32_bf16<<<(n + 255) / 256, 256, 0, stream>>>(src, dst, n);
  };
  cast(w_qkv,  wqkv_bf,  D_MODEL * QKVN);
  cast(w_out,  wout_bf,  D_MODEL * D_MODEL);
  cast(w_ffn1, wffn1_bf, D_MODEL * DFF);
  cast(w_ffn2, wffn2_bf, DFF * D_MODEL);

  // 1) h = rmsnorm(x) (bf16)
  rmsnorm_kernel<<<MROWS, 256, 0, stream>>>(x, norm1_w, hA);

  // 2) qkv = h @ w_qkv (bf16 out)
  gemm_bf16_kernel<3><<<dim3(QKVN / 128, MROWS / 128), 256, 0, stream>>>(
      hA, wqkv_bf, nullptr, qkv_bf, nullptr, nullptr, MROWS, QKVN, D_MODEL);

  // 3) sliding-window GQA attention (bf16 out)
  attn_kernel<<<dim3(S_LEN / 16, HKV, B_SZ), 128, 0, stream>>>(qkv_bf, attn_bf);

  // 4) x1 = x + attn @ w_out (f32)
  gemm_bf16_kernel<0><<<dim3(D_MODEL / 128, MROWS / 128), 256, 0, stream>>>(
      attn_bf, wout_bf, x1, nullptr, nullptr, x, MROWS, D_MODEL, D_MODEL);

  // 5) h2 = rmsnorm(x1) (bf16)
  rmsnorm_kernel<<<MROWS, 256, 0, stream>>>(x1, norm2_w, h2);

  // 6) ffn1 = gelu(h2 @ w_ffn1 + b1) (bf16)
  gemm_bf16_kernel<1><<<dim3(DFF / 128, MROWS / 128), 256, 0, stream>>>(
      h2, wffn1_bf, nullptr, ffn1_bf, b_ffn1, nullptr, MROWS, DFF, D_MODEL);

  // 7) out = x1 + ffn1 @ w_ffn2 + b2 (f32 -> d_out)
  gemm_bf16_kernel<2><<<dim3(D_MODEL / 128, MROWS / 128), 256, 0, stream>>>(
      ffn1_bf, wffn2_bf, outp, nullptr, b_ffn2, x1, MROWS, D_MODEL, DFF);
}
